// MovementInjector_1855425872135
// MI455X (gfx1250) — matrix operand order repaired and checked by reading, not compiled, not run on hardware
//
#include <hip/hip_runtime.h>
#include <hip/hip_bf16.h>

// ---------------- problem constants (from reference) ----------------
#define T_FR    3
#define S_TOK   3520          // 44*80
#define M_TOK   10560         // T*S
#define C_IMG   3072
#define KH      1024
#define NH      16
#define HD      64
#define HID     512
#define KD_IN   16
#define L_KV    3
#define EPSF    1e-6f

typedef __attribute__((ext_vector_type(16))) __bf16 bf16x16;
typedef __attribute__((ext_vector_type(8)))  float  floatx8;

union FragAB { uint4 u[2]; bf16x16 v; };

static __device__ __forceinline__ unsigned short f2bf(float f) {
    unsigned int u = __float_as_uint(f);
    unsigned int r = u + 0x7FFFu + ((u >> 16) & 1u);   // round-to-nearest-even
    return (unsigned short)(r >> 16);
}

// ---- CDNA5 async global->LDS copy (ASYNCcnt-tracked DMA, 16B per lane) ----
static __device__ __forceinline__ void async_copy_b128(unsigned lds_addr,
                                                       const unsigned short* g) {
    asm volatile("global_load_async_to_lds_b128 %0, %1, off"
                 :: "v"(lds_addr), "v"((const void*)g)
                 : "memory");
}
static __device__ __forceinline__ void wait_async0() {
#if __has_builtin(__builtin_amdgcn_s_wait_asynccnt)
    __builtin_amdgcn_s_wait_asynccnt(0);
#else
    asm volatile("s_wait_asynccnt 0x0" ::: "memory");
#endif
}

// ---------------- f32 -> bf16 conversion ----------------
__global__ void cvt_bf16_kernel(const float* __restrict__ in,
                                unsigned short* __restrict__ out, size_t n) {
    size_t i = (size_t)blockIdx.x * blockDim.x + threadIdx.x;
    size_t stride = (size_t)gridDim.x * blockDim.x;
    for (; i < n; i += stride) out[i] = f2bf(in[i]);
}

// ---------------- f32 (KxN) -> bf16 transposed (NxK) ----------------
__global__ void transpose_cvt_kernel(const float* __restrict__ W,
                                     unsigned short* __restrict__ Wt,
                                     int K, int N) {
    __shared__ float tile[32][33];
    const int tx = threadIdx.x;          // 0..31
    const int ty = threadIdx.y;          // 0..7
    const int k0 = blockIdx.y * 32;
    const int n0 = blockIdx.x * 32;
    #pragma unroll
    for (int rr = ty; rr < 32; rr += 8)
        tile[rr][tx] = W[(size_t)(k0 + rr) * N + (n0 + tx)];
    __syncthreads();
    #pragma unroll
    for (int rr = ty; rr < 32; rr += 8)
        Wt[(size_t)(n0 + rr) * K + (k0 + tx)] = f2bf(tile[tx][rr]);
}

// ---------------- cond MLP + window gather -> mf[3][4096] ----------------
__global__ void preprocess_kernel(const float* __restrict__ cond,
                                  const float* __restrict__ w1, const float* __restrict__ b1,
                                  const float* __restrict__ w2, const float* __restrict__ b2,
                                  float* __restrict__ mf) {
    __shared__ float cpr[KD_IN];
    __shared__ float h1[HID];
    const int rr  = blockIdx.x;      // 0..15 -> emb row 72+rr -> cond row 64+rr
    const int tid = threadIdx.x;
    if (tid < KD_IN) cpr[tid] = cond[(64 + rr) * KD_IN + tid];
    __syncthreads();
    #pragma unroll
    for (int p = 0; p < 2; ++p) {
        int j = tid + p * 256;
        float s = b1[j];
        #pragma unroll
        for (int i = 0; i < KD_IN; ++i) s += cpr[i] * w1[i * HID + j];
        h1[j] = s / (1.0f + __expf(-s));     // silu
    }
    __syncthreads();
    #pragma unroll
    for (int p = 0; p < 2; ++p) {
        int j = tid + p * 256;
        float e = b2[j];
        for (int i = 0; i < HID; ++i) e += h1[i] * w2[i * HID + j];
        #pragma unroll
        for (int t = 0; t < L_KV; ++t) {     // mf[t][jj*512+j], rr = t*4 + jj
            int jj = rr - 4 * t;
            if (jj >= 0 && jj < 8) mf[(size_t)t * 4096 + jj * HID + j] = e;
        }
    }
}

// ---------------- kv = mf @ kv_w + kv_b -> k_raw / v ----------------
__global__ void kv_kernel(const float* __restrict__ mf,
                          const float* __restrict__ kvw, const float* __restrict__ kvb,
                          float* __restrict__ k_raw, float* __restrict__ v) {
    __shared__ float mfs[4096];
    const int t   = blockIdx.y;
    const int col = blockIdx.x * 256 + threadIdx.x;
    for (int i = threadIdx.x; i < 4096; i += 256) mfs[i] = mf[(size_t)t * 4096 + i];
    __syncthreads();
    float acc = kvb[col];
    for (int i = 0; i < 4096; ++i) acc += mfs[i] * kvw[(size_t)i * 2048 + col];
    if (col < KH) k_raw[t * KH + col] = acc;
    else          v[t * KH + (col - KH)] = acc;
}

// ---------------- rmsnorm(k) over HD per (t,h) ----------------
__global__ void knorm_kernel(const float* __restrict__ k_raw,
                             const float* __restrict__ knw,
                             float* __restrict__ k_norm) {
    __shared__ float red[HD];
    const int row = blockIdx.x;          // 0..47
    const int d   = threadIdx.x;         // 0..63
    float x = k_raw[row * HD + d];
    red[d] = x * x;
    __syncthreads();
    for (int s = 32; s > 0; s >>= 1) {
        if (d < s) red[d] += red[d + s];
        __syncthreads();
    }
    float rs = rsqrtf(red[0] * (1.0f / HD) + EPSF);
    k_norm[row * HD + d] = x * rs * knw[d];
}

// ---------------- tiled bf16 WMMA GEMM: C = A(MxK) * Bt(NxK)^T + bias (+res)
// BM=128, BN=128, BK=32; 8 waves (4x2), each wave 32x64 (2x4 WMMA tiles).
// Double-buffered LDS, async global->LDS staging, one barrier per K-step.
#define LDA 40
#define LDB 40

template <bool RES>
__global__ __launch_bounds__(256)
void gemm_bf16_kernel(const unsigned short* __restrict__ A,    // M x K
                      const unsigned short* __restrict__ Bt,   // N x K (pre-transposed)
                      const float* __restrict__ bias,
                      const float* __restrict__ residual,
                      float* __restrict__ C,
                      int M, int N, int K) {
    __shared__ __align__(16) unsigned short As[2][128 * LDA];
    __shared__ __align__(16) unsigned short Bs[2][128 * LDB];

    const int tid  = threadIdx.x;
    const int lane = tid & 31;
    const int wave = tid >> 5;
    const int r = lane & 15, h = lane >> 4;
    const int wm = wave >> 1;            // 0..3 -> row offset wm*32
    const int wn = wave & 1;             // 0..1 -> col offset wn*64

    const int m0 = blockIdx.y * 128;
    const int n0 = blockIdx.x * 128;

    // staging: 128 rows x 32 K, 32B (two b128) per thread, same map for A and B
    const int srow = tid >> 1;
    const int shalf = tid & 1;
    const bool a_ok = (m0 + srow) < M;
    const unsigned short* aSrc = A  + (size_t)(m0 + srow) * K + shalf * 16;
    const unsigned short* bSrc = Bt + (size_t)(n0 + srow) * K + shalf * 16;
    const unsigned aLds[2] = {
        (unsigned)(size_t)&As[0][srow * LDA + shalf * 16],
        (unsigned)(size_t)&As[1][srow * LDA + shalf * 16] };
    const unsigned bLds[2] = {
        (unsigned)(size_t)&Bs[0][srow * LDB + shalf * 16],
        (unsigned)(size_t)&Bs[1][srow * LDB + shalf * 16] };

    floatx8 acc[2][4];
    #pragma unroll
    for (int mi = 0; mi < 2; ++mi)
        #pragma unroll
        for (int ni = 0; ni < 4; ++ni) acc[mi][ni] = (floatx8)0.0f;

    const int nK = K >> 5;

    auto stage = [&](int kt, int buf) {
        const size_t koff = (size_t)kt * 32;
        if (a_ok) {
            async_copy_b128(aLds[buf],      aSrc + koff);
            async_copy_b128(aLds[buf] + 16, aSrc + koff + 8);
        } else {
            uint4 z = {0u, 0u, 0u, 0u};
            uint4* dz = (uint4*)&As[buf][srow * LDA + shalf * 16];
            dz[0] = z; dz[1] = z;
        }
        async_copy_b128(bLds[buf],      bSrc + koff);
        async_copy_b128(bLds[buf] + 16, bSrc + koff + 8);
    };

    auto compute = [&](int buf) {
        FragAB a[2], b[4];
        #pragma unroll
        for (int mi = 0; mi < 2; ++mi) {
            const int row = wm * 32 + mi * 16 + r;
            const uint4* p = (const uint4*)&As[buf][row * LDA + h * 8];
            a[mi].u[0] = p[0];     // K = h*8 .. h*8+7
            a[mi].u[1] = p[2];     // K = h*8+16 .. h*8+23
        }
        #pragma unroll
        for (int ni = 0; ni < 4; ++ni) {
            const int col = wn * 64 + ni * 16 + r;
            const uint4* p = (const uint4*)&Bs[buf][col * LDB + h * 8];
            b[ni].u[0] = p[0];     // K = h*8 .. h*8+7
            b[ni].u[1] = p[2];     // K = h*8+16 .. h*8+23
        }
        #pragma unroll
        for (int mi = 0; mi < 2; ++mi)
            #pragma unroll
            for (int ni = 0; ni < 4; ++ni)
                acc[mi][ni] = __builtin_amdgcn_wmma_f32_16x16x32_bf16(
                    false, a[mi].v, false, b[ni].v, (short)0, acc[mi][ni], false, false);
    };

    stage(0, 0);
    wait_async0();
    __syncthreads();

    for (int kt = 0; kt < nK; ++kt) {
        const int cur = kt & 1;
        if (kt + 1 < nK) stage(kt + 1, cur ^ 1);
        if (kt + 2 < nK && a_ok)
            __builtin_prefetch(aSrc + (size_t)(kt + 2) * 32, 0, 1);
        compute(cur);
        if (kt + 1 < nK) {
            wait_async0();
            __syncthreads();
        }
    }

    // ---- epilogue: bias (+ residual), f32 store ----
    #pragma unroll
    for (int mi = 0; mi < 2; ++mi) {
        #pragma unroll
        for (int ni = 0; ni < 4; ++ni) {
            const int col = n0 + wn * 64 + ni * 16 + r;
            const float bcol = bias[col];
            #pragma unroll
            for (int i = 0; i < 8; ++i) {
                const int row = m0 + wm * 32 + mi * 16 + h * 8 + i;
                if (row < M) {
                    float val = acc[mi][ni][i] + bcol;
                    if (RES) val += residual[(size_t)row * N + col];
                    C[(size_t)row * N + col] = val;
                }
            }
        }
    }
}

// ---------------- attention: rmsnorm(q), softmax over L=3, weighted V ------
__global__ void attn_kernel(const float* __restrict__ Qraw,
                            const float* __restrict__ qnw,
                            const float* __restrict__ k_norm,
                            const float* __restrict__ v,
                            unsigned short* __restrict__ Obf) {
    const int gid = blockIdx.x * blockDim.x + threadIdx.x;
    if (gid >= M_TOK * NH) return;
    const int rid = gid >> 4;       // token row (t*S+s)
    const int hh  = gid & 15;       // head
    const float4* q4 = (const float4*)(Qraw + (size_t)rid * KH + hh * HD);
    const float4* w4 = (const float4*)qnw;

    float ssq = 0.0f;
    float4 qv[16];
    #pragma unroll
    for (int i = 0; i < 16; ++i) {
        float4 t = q4[i];
        qv[i] = t;
        ssq += t.x * t.x + t.y * t.y + t.z * t.z + t.w * t.w;
    }
    const float rs = rsqrtf(ssq * (1.0f / HD) + EPSF);

    float sc[L_KV];
    #pragma unroll
    for (int l = 0; l < L_KV; ++l) {
        const float4* kb = (const float4*)(k_norm + (l * NH + hh) * HD);
        float d = 0.0f;
        #pragma unroll
        for (int i = 0; i < 16; ++i) {
            float4 qq = qv[i], ww = w4[i], kk = kb[i];
            d += qq.x * ww.x * kk.x + qq.y * ww.y * kk.y +
                 qq.z * ww.z * kk.z + qq.w * ww.w * kk.w;
        }
        sc[l] = d * rs * 0.125f;    // 1/sqrt(64)
    }
    float mx = fmaxf(sc[0], fmaxf(sc[1], sc[2]));
    float p0 = __expf(sc[0] - mx), p1 = __expf(sc[1] - mx), p2 = __expf(sc[2] - mx);
    const float inv = 1.0f / (p0 + p1 + p2);
    p0 *= inv; p1 *= inv; p2 *= inv;

    const float4* v0 = (const float4*)(v + (0 * NH + hh) * HD);
    const float4* v1 = (const float4*)(v + (1 * NH + hh) * HD);
    const float4* v2 = (const float4*)(v + (2 * NH + hh) * HD);
    unsigned short* ob = Obf + (size_t)rid * KH + hh * HD;
    #pragma unroll
    for (int i = 0; i < 16; ++i) {
        float4 a = v0[i], b = v1[i], c = v2[i];
        float ox = p0 * a.x + p1 * b.x + p2 * c.x;
        float oy = p0 * a.y + p1 * b.y + p2 * c.y;
        float oz = p0 * a.z + p1 * b.z + p2 * c.z;
        float ow = p0 * a.w + p1 * b.w + p2 * c.w;
        uint2 packed;
        packed.x = (unsigned)f2bf(ox) | ((unsigned)f2bf(oy) << 16);
        packed.y = (unsigned)f2bf(oz) | ((unsigned)f2bf(ow) << 16);
        *(uint2*)&ob[i * 4] = packed;
    }
}

// ---------------- host-side orchestration ----------------
extern "C" void kernel_launch(void* const* d_in, const int* in_sizes, int n_in,
                              void* d_out, int out_size, void* d_ws, size_t ws_size,
                              hipStream_t stream) {
    const float* x     = (const float*)d_in[0];
    const float* cond  = (const float*)d_in[1];
    const float* me_w1 = (const float*)d_in[2];
    const float* me_b1 = (const float*)d_in[3];
    const float* me_w2 = (const float*)d_in[4];
    const float* me_b2 = (const float*)d_in[5];
    const float* q_w   = (const float*)d_in[6];
    const float* q_b   = (const float*)d_in[7];
    const float* kv_w  = (const float*)d_in[8];
    const float* kv_b  = (const float*)d_in[9];
    const float* qn_w  = (const float*)d_in[10];
    const float* kn_w  = (const float*)d_in[11];
    const float* out_w = (const float*)d_in[12];
    const float* out_b = (const float*)d_in[13];
    float* out = (float*)d_out;

    char* p = (char*)d_ws;
    auto alloc = [&](size_t bytes) -> void* {
        void* rv = (void*)p;
        p += (bytes + 255) & ~(size_t)255;
        return rv;
    };
    unsigned short* x_bf   = (unsigned short*)alloc((size_t)M_TOK * C_IMG * 2);
    unsigned short* qw_t   = (unsigned short*)alloc((size_t)KH * C_IMG * 2);    // 1024 x 3072
    unsigned short* ow_t   = (unsigned short*)alloc((size_t)C_IMG * KH * 2);    // 3072 x 1024
    float*          Qraw   = (float*)alloc((size_t)M_TOK * KH * 4);
    unsigned short* O_bf   = (unsigned short*)alloc((size_t)M_TOK * KH * 2);
    float*          mf     = (float*)alloc((size_t)L_KV * 4096 * 4);
    float*          k_raw  = (float*)alloc((size_t)L_KV * KH * 4);
    float*          k_norm = (float*)alloc((size_t)L_KV * KH * 4);
    float*          v_buf  = (float*)alloc((size_t)L_KV * KH * 4);

    // 1) conversions: x -> bf16 ; weights -> bf16 transposed (N-major)
    cvt_bf16_kernel<<<4096, 256, 0, stream>>>(x, x_bf, (size_t)M_TOK * C_IMG);
    transpose_cvt_kernel<<<dim3(KH / 32, C_IMG / 32), dim3(32, 8), 0, stream>>>(
        q_w, qw_t, C_IMG, KH);
    transpose_cvt_kernel<<<dim3(C_IMG / 32, KH / 32), dim3(32, 8), 0, stream>>>(
        out_w, ow_t, KH, C_IMG);

    // 2) tiny path: cond MLP -> mf ; kv proj ; k rmsnorm
    preprocess_kernel<<<16, 256, 0, stream>>>(cond, me_w1, me_b1, me_w2, me_b2, mf);
    kv_kernel<<<dim3(8, 3), 256, 0, stream>>>(mf, kv_w, kv_b, k_raw, v_buf);
    knorm_kernel<<<48, 64, 0, stream>>>(k_raw, kn_w, k_norm);

    // 3) Q projection: Qraw = x @ q_w + q_b
    gemm_bf16_kernel<false><<<dim3(KH / 128, (M_TOK + 127) / 128), 256, 0, stream>>>(
        x_bf, qw_t, q_b, nullptr, Qraw, M_TOK, KH, C_IMG);

    // 4) attention (L=3 keys) -> O in bf16
    attn_kernel<<<(M_TOK * NH + 255) / 256, 256, 0, stream>>>(Qraw, qn_w, k_norm, v_buf, O_bf);

    // 5) out projection + residual: out = x + O @ out_w + out_b
    gemm_bf16_kernel<true><<<dim3(C_IMG / 128, (M_TOK + 127) / 128), 256, 0, stream>>>(
        O_bf, ow_t, out_b, x, out, M_TOK, C_IMG, KH);

    (void)in_sizes; (void)n_in; (void)out_size; (void)ws_size;
}
